// NCN_6545530159541
// MI455X (gfx1250) — matrix operand order repaired and checked by reading, not compiled, not run on hardware
//
#include <hip/hip_runtime.h>
#include <hip/hip_bf16.h>
#include <stdint.h>

#define N_NODES 16384
#define N_LINKS 8192
#define D_EMB   128

// CDNA5 WMMA vector types (wave32): 16 bf16 = 8 VGPRs, 8 f32 = 8 VGPRs
typedef __attribute__((ext_vector_type(16))) __bf16 v16bf;
typedef __attribute__((ext_vector_type(8)))  float  v8f;

union BF16x16 { uint32_t u[8]; v16bf v; };
union F32x8   { float    f[8]; v8f  v; };

// byte pair {b_lo, b_hi} (each 0/1) -> packed bf16 pair {1.0/0.0, 1.0/0.0}
// v_perm_b32 spreads the two bytes to byte lanes 0 and 2 (sel 0x0C == 0x00),
// then a single v_mul by 0x3F80 (bf16 1.0) builds both halves.
__device__ __forceinline__ uint32_t mask2_to_bf16(uint32_t hi32, uint32_t lo32,
                                                  uint32_t sel) {
    return __builtin_amdgcn_perm(hi32, lo32, sel) * 0x3F80u;
}

// ---------------------------------------------------------------------------
// Kernel 1: out[l, 0:128] = emb[i]*emb[j]  (elementwise product half)
// ---------------------------------------------------------------------------
__global__ void ncn_product_kernel(const int* __restrict__ links,
                                   const float* __restrict__ emb,
                                   float* __restrict__ out) {
    const int l = blockIdx.x;      // link
    const int d = threadIdx.x;     // 0..127
    const int i = links[2 * l];
    const int j = links[2 * l + 1];
    out[(size_t)l * 256 + d] = emb[(size_t)i * D_EMB + d] * emb[(size_t)j * D_EMB + d];
}

// ---------------------------------------------------------------------------
// Kernel 2: embT[d][n] = bf16(emb[n][d])   (transpose + convert, 4MB in d_ws)
// Makes WMMA B-matrix loads contiguous along K (node index).
// ---------------------------------------------------------------------------
__global__ void ncn_embT_kernel(const float* __restrict__ emb,
                                __hip_bfloat16* __restrict__ embT) {
    const int n = blockIdx.x;      // node
    const int d = threadIdx.x;     // dim 0..127
    embT[(size_t)d * N_NODES + n] = __float2bfloat16(emb[(size_t)n * D_EMB + d]);
}

// ---------------------------------------------------------------------------
// Kernel 3: cn_sum via V_WMMA_F32_16X16X32_BF16.
// Block: 16 links x 128 dims. 8 waves, wave w owns K-partition [w*2048, w*2048+2048).
// Per 32-K chunk: issue ALL vmem reads (4 adjacency b64 + 16 embT b128) as one
// scheduling group, build A (16x32 bf16 mask) with perm+mul, then burst 8 WMMAs
// with staged loadcnt waits. Cross-wave K-reduction through LDS at the end.
// ---------------------------------------------------------------------------
__global__ void __launch_bounds__(256, 1)
ncn_cnsum_wmma_kernel(const int* __restrict__ links,
                      const uint8_t* __restrict__ adj,         // bool rows, 1 byte/elem
                      const __hip_bfloat16* __restrict__ embT, // [128][16384] bf16
                      float* __restrict__ out) {
    __shared__ float red[8 * 8 * 32 * 8];   // [wave][ntile][lane][8] = 64 KB

    const int lane = threadIdx.x & 31;
    const int wave = threadIdx.x >> 5;      // 0..7 = K partition
    const int half = lane >> 4;             // 0 or 1 (lane group)
    const int row  = lane & 15;             // M index / N index within tile

    const int linkBase = blockIdx.x * 16;
    const int link  = linkBase + row;
    const int iNode = links[2 * link];
    const int jNode = links[2 * link + 1];
    const uint8_t* adjI = adj + (size_t)iNode * N_NODES;
    const uint8_t* adjJ = adj + (size_t)jNode * N_NODES;

    const int kPart = wave * (N_NODES / 8); // 2048 nodes per wave

    F32x8 acc[8];
#pragma unroll
    for (int nt = 0; nt < 8; ++nt)
#pragma unroll
        for (int r = 0; r < 8; ++r) acc[nt].f[r] = 0.0f;

    // B base pointer: this lane supplies column N = (lane&15) of each N-tile.
    const __hip_bfloat16* bRow = embT + (size_t)row * N_NODES;

    for (int kc = 0; kc < N_NODES / 8; kc += 32) {
        // ---- A source: adjacency mask bytes, ISA A layout:
        // lanes 0-15 : V0-3 = K 0..7,  V4-7 = K 16..23   (kbA = base+0,  +16)
        // lanes 16-31: V0-3 = K 8..15, V4-7 = K 24..31   (kbA = base+8,  +24)
        const int kbA = kPart + kc + half * 8;
        const uint64_t mLo = (*(const uint64_t*)(adjI + kbA)) &
                             (*(const uint64_t*)(adjJ + kbA));
        const uint64_t mHi = (*(const uint64_t*)(adjI + kbA + 16)) &
                             (*(const uint64_t*)(adjJ + kbA + 16));

        // ---- B tiles: all 8 N-tiles up front.
        // B layout (32x16 bf16): lanes 0-15: K 0..7,16..23; lanes 16-31: K 8..15,24..31
        const int kbB = kPart + kc + half * 8;
        BF16x16 Bt[8];
#pragma unroll
        for (int nt = 0; nt < 8; ++nt) {
            const __hip_bfloat16* bp = bRow + (size_t)nt * 16 * N_NODES + kbB;
            const uint4 lo = *(const uint4*)(const void*)(bp);      // K +0..7
            const uint4 hi = *(const uint4*)(const void*)(bp + 16); // K +16..23
            Bt[nt].u[0] = lo.x; Bt[nt].u[1] = lo.y;
            Bt[nt].u[2] = lo.z; Bt[nt].u[3] = lo.w;
            Bt[nt].u[4] = hi.x; Bt[nt].u[5] = hi.y;
            Bt[nt].u[6] = hi.z; Bt[nt].u[7] = hi.w;
        }

        // ---- mask bytes {0,1} -> packed bf16 {0.0,1.0} pairs: 2 VALU each
        const uint32_t l0 = (uint32_t)mLo, l1 = (uint32_t)(mLo >> 32);
        const uint32_t h0 = (uint32_t)mHi, h1 = (uint32_t)(mHi >> 32);
        BF16x16 A;
        A.u[0] = mask2_to_bf16(l1, l0, 0x0C010C00u);  // bytes 0,1
        A.u[1] = mask2_to_bf16(l1, l0, 0x0C030C02u);  // bytes 2,3
        A.u[2] = mask2_to_bf16(l1, l0, 0x0C050C04u);  // bytes 4,5
        A.u[3] = mask2_to_bf16(l1, l0, 0x0C070C06u);  // bytes 6,7
        A.u[4] = mask2_to_bf16(h1, h0, 0x0C010C00u);
        A.u[5] = mask2_to_bf16(h1, h0, 0x0C030C02u);
        A.u[6] = mask2_to_bf16(h1, h0, 0x0C050C04u);
        A.u[7] = mask2_to_bf16(h1, h0, 0x0C070C06u);

        // ---- WMMA burst: one A reused across 8 N-tiles
#pragma unroll
        for (int nt = 0; nt < 8; ++nt) {
            acc[nt].v = __builtin_amdgcn_wmma_f32_16x16x32_bf16(
                false, A.v, false, Bt[nt].v, (short)0, acc[nt].v, false, false);
        }

        // ---- Scheduling pins (IGroupLP): per iteration, issue every VMEM
        // read first (4 adjacency + 16 B -> one deep clause, staged waits),
        // then the A-build VALU, then the 8 WMMAs.
        __builtin_amdgcn_sched_group_barrier(0x020, 20, 0); // VMEM reads
        __builtin_amdgcn_sched_group_barrier(0x002, 32, 0); // VALU (A build, addr)
        __builtin_amdgcn_sched_group_barrier(0x008, 8, 0);  // WMMA burst
    }

    // ---- cross-wave reduction over K partitions
#pragma unroll
    for (int nt = 0; nt < 8; ++nt)
#pragma unroll
        for (int r = 0; r < 8; ++r)
            red[(((wave * 8 + nt) * 32) + lane) * 8 + r] = acc[nt].f[r];

    __syncthreads();

    // wave w reduces N-tile nt == w across all 8 K partitions
    float s[8];
#pragma unroll
    for (int r = 0; r < 8; ++r) s[r] = 0.0f;
#pragma unroll
    for (int wv = 0; wv < 8; ++wv)
#pragma unroll
        for (int r = 0; r < 8; ++r)
            s[r] += red[(((wv * 8 + wave) * 32) + lane) * 8 + r];

    // C/D layout: N = lane&15, M = half*8 + r  -> out[link, 128 + wave*16 + N]
#pragma unroll
    for (int r = 0; r < 8; ++r) {
        const int M = half * 8 + r;
        out[(size_t)(linkBase + M) * 256 + 128 + wave * 16 + (lane & 15)] = s[r];
    }
}

// ---------------------------------------------------------------------------
extern "C" void kernel_launch(void* const* d_in, const int* in_sizes, int n_in,
                              void* d_out, int out_size, void* d_ws, size_t ws_size,
                              hipStream_t stream) {
    const int*      links = (const int*)d_in[0];        // targetLink [8192][2] i32
    const uint8_t*  adj   = (const uint8_t*)d_in[1];    // adjacent [16384][16384] bool
    const float*    emb   = (const float*)d_in[2];      // NodeEmbedding [16384][128] f32
    float*          out   = (float*)d_out;              // [8192][256] f32
    __hip_bfloat16* embT  = (__hip_bfloat16*)d_ws;      // [128][16384] bf16 (4 MB)

    ncn_product_kernel<<<N_LINKS, D_EMB, 0, stream>>>(links, emb, out);
    ncn_embT_kernel<<<N_NODES, D_EMB, 0, stream>>>(emb, embT);
    ncn_cnsum_wmma_kernel<<<N_LINKS / 16, 256, 0, stream>>>(links, adj, embT, out);
}
